// DiTAttention_23922967839279
// MI455X (gfx1250) — matrix operand order repaired and checked by reading, not compiled, not run on hardware
//
#include <hip/hip_runtime.h>
#include <hip/hip_bf16.h>

#define DEVINL __device__ __forceinline__

typedef __attribute__((ext_vector_type(16))) __bf16 v16bf;
typedef __attribute__((ext_vector_type(8)))  __bf16 v8bf;
typedef __attribute__((ext_vector_type(4)))  __bf16 v4bf;
typedef __attribute__((ext_vector_type(8)))  float  v8f;
typedef __attribute__((ext_vector_type(4)))  float  v4f;

constexpr int BATCH = 2;
constexpr int SEQL  = 2048;         // T*GH*GW
constexpr int NH    = 24;
constexpr int DH    = 64;
constexpr int HID   = 1536;
constexpr int EQKV  = 3 * NH * DH;  // 4608
constexpr int MROWS = BATCH * SEQL; // 4096

// ---------------------------------------------------------------------------
// WMMA fragment helpers (layouts per CDNA5 ISA 7.12.2)
// A 16x32 bf16: lane = m + 16*h  holds K in [8h,8h+8) and [16+8h,16+8h+8)
// B 32x16 bf16: lane = n + 16*h  holds K in [8h,8h+8) and [16+8h,16+8h+8)
// C/D 16x16 f32: VGPR i, lane = n + 16*h -> element (m = i + 8h, n)
// ---------------------------------------------------------------------------
DEVINL v16bf load_a_frag(const __bf16* __restrict__ base, int ld, int lane) {
    const int m = lane & 15, h = lane >> 4;
    const __bf16* p = base + (size_t)m * ld + 8 * h;
    v8bf c0 = *(const v8bf*)(p);
    v8bf c1 = *(const v8bf*)(p + 16);
    v16bf f;
#pragma unroll
    for (int i = 0; i < 8; ++i) { f[i] = c0[i]; f[i + 8] = c1[i]; }
    return f;
}

DEVINL v16bf load_b_frag(const __bf16* __restrict__ base, int ld, int lane) {
    const int n = lane & 15, h = lane >> 4;
    const __bf16* p = base + (size_t)n * ld + 8 * h;
    v8bf c0 = *(const v8bf*)(p);
    v8bf c1 = *(const v8bf*)(p + 16);
    v16bf f;
#pragma unroll
    for (int i = 0; i < 8; ++i) { f[i] = c0[i]; f[i + 8] = c1[i]; }
    return f;
}

DEVINL v8f wmma_bf16(v16bf a, v16bf b, v8f c) {
    return __builtin_amdgcn_wmma_f32_16x16x32_bf16(false, a, false, b,
                                                   (short)0, c, false, false);
}

// ---------------------------------------------------------------------------
// fp32 -> bf16 conversion, 4 elements/thread (b128 load -> b64 store)
// ---------------------------------------------------------------------------
__global__ void cvt_bf16x4_kernel(const float* __restrict__ in,
                                  __bf16* __restrict__ out, int n4) {
    int i = blockIdx.x * blockDim.x + threadIdx.x;
    if (i < n4) {
        v4f v = *(const v4f*)(in + 4 * (size_t)i);
        v4bf b;
#pragma unroll
        for (int j = 0; j < 4; ++j) b[j] = (__bf16)v[j];
        *(v4bf*)(out + 4 * (size_t)i) = b;
    }
}

// ---------------------------------------------------------------------------
// GEMM: C[m,n] = sum_k A[m,k] * W[n,k]   (both operands K-contiguous)
// Wave tile 64x32 (4 M-tiles x 2 N-tiles): 8 WMMAs per 12 b128 loads.
// Block = 8 waves (2x4) covering a 128x128 tile (better L2 panel reuse).
// ---------------------------------------------------------------------------
template <bool OUT_BF16>
__global__ __launch_bounds__(256) void gemm_bf16_kernel(
    const __bf16* __restrict__ A, const __bf16* __restrict__ W,
    void* __restrict__ Cout, int M, int N, int K) {
    const int lane = threadIdx.x & 31;
    const int wave = threadIdx.x >> 5;
    const int m0 = blockIdx.y * 128 + (wave >> 2) * 64;
    const int n0 = blockIdx.x * 128 + (wave & 3) * 32;

    v8f acc[4][2];
#pragma unroll
    for (int mi = 0; mi < 4; ++mi)
#pragma unroll
        for (int ni = 0; ni < 2; ++ni) acc[mi][ni] = (v8f){};

#pragma unroll 2
    for (int kb = 0; kb < K; kb += 32) {
        v16bf a0 = load_a_frag(A + (size_t)(m0 +  0) * K + kb, K, lane);
        v16bf a1 = load_a_frag(A + (size_t)(m0 + 16) * K + kb, K, lane);
        v16bf a2 = load_a_frag(A + (size_t)(m0 + 32) * K + kb, K, lane);
        v16bf a3 = load_a_frag(A + (size_t)(m0 + 48) * K + kb, K, lane);
        v16bf b0 = load_b_frag(W + (size_t)(n0 +  0) * K + kb, K, lane);
        v16bf b1 = load_b_frag(W + (size_t)(n0 + 16) * K + kb, K, lane);
        acc[0][0] = wmma_bf16(a0, b0, acc[0][0]);
        acc[0][1] = wmma_bf16(a0, b1, acc[0][1]);
        acc[1][0] = wmma_bf16(a1, b0, acc[1][0]);
        acc[1][1] = wmma_bf16(a1, b1, acc[1][1]);
        acc[2][0] = wmma_bf16(a2, b0, acc[2][0]);
        acc[2][1] = wmma_bf16(a2, b1, acc[2][1]);
        acc[3][0] = wmma_bf16(a3, b0, acc[3][0]);
        acc[3][1] = wmma_bf16(a3, b1, acc[3][1]);
    }
    const int h = lane >> 4, cn = lane & 15;
#pragma unroll
    for (int mi = 0; mi < 4; ++mi) {
#pragma unroll
        for (int ni = 0; ni < 2; ++ni) {
#pragma unroll
            for (int i = 0; i < 8; ++i) {
                int r = m0 + mi * 16 + i + 8 * h;
                int c = n0 + ni * 16 + cn;
                if (OUT_BF16)
                    ((__bf16*)Cout)[(size_t)r * N + c] = (__bf16)acc[mi][ni][i];
                else
                    ((float*)Cout)[(size_t)r * N + c] = acc[mi][ni][i];
            }
        }
    }
}

// ---------------------------------------------------------------------------
// RoPE + head split:
//   qkv[b,l, (s*NH+h)*64+d]  (bf16)  ->
//   Q[b,h,l,d] (scaled by 1/sqrt(64)), K[b,h,l,d], Vt[b,h,d,l]
// ---------------------------------------------------------------------------
__global__ __launch_bounds__(256) void rope_split_kernel(
    const __bf16* __restrict__ qkv, const float* __restrict__ cosb,
    const float* __restrict__ sinb, __bf16* __restrict__ Q,
    __bf16* __restrict__ Kd, __bf16* __restrict__ Vt) {
    int idx = blockIdx.x * blockDim.x + threadIdx.x; // over B*L*NH*DH
    int d  = idx & 63;
    int hh = (idx >> 6) % NH;
    int l  = (idx >> 6) / NH % SEQL;
    int b  = idx / (64 * NH * SEQL);

    const __bf16* row = qkv + (size_t)(b * SEQL + l) * EQKV;
    float c = cosb[l * 64 + d];
    float s = sinb[l * 64 + d];
    int d2 = (d + 32) & 63;
    float sign = (d < 32) ? -1.0f : 1.0f;

    float qv = (float)row[hh * 64 + d];
    float qr = (float)row[hh * 64 + d2];
    float kv = (float)row[(NH + hh) * 64 + d];
    float kr = (float)row[(NH + hh) * 64 + d2];
    float vv = (float)row[(2 * NH + hh) * 64 + d];

    float q2 = qv * c + sign * qr * s;
    float k2 = kv * c + sign * kr * s;

    size_t o = ((size_t)(b * NH + hh) * SEQL + l) * 64 + d;
    Q[o]  = (__bf16)(q2 * 0.125f); // fold in 1/sqrt(HEAD_DIM)
    Kd[o] = (__bf16)k2;
    Vt[((size_t)(b * NH + hh) * 64 + d) * SEQL + l] = (__bf16)vv;
}

// ---------------------------------------------------------------------------
// Flash attention, transposed-S formulation. One wave per (b,h, 32 queries);
// the 4 waves of a block handle consecutive query tiles of the SAME head so
// their K/V streams coalesce in the WGP cache.
// S^T = K Q^T: C-layout of S^T (M=key, N=query) is *identical* to the
// A-fragment layout of P for the P@V WMMA -> P never leaves registers.
// Per 32-key step: 16 WMMAs, 16 b128 loads, 1 shfl_xor(16) per reduction.
// Q[b,h,l,d], K[b,h,l,d], Vt[b,h,d,l] -> attn_out[b,l, h*64+d] (bf16)
// ---------------------------------------------------------------------------
__global__ __launch_bounds__(128) void flash_attn_kernel(
    const __bf16* __restrict__ Q, const __bf16* __restrict__ Kd,
    const __bf16* __restrict__ Vt, __bf16* __restrict__ Oa) {
    const int lane = threadIdx.x & 31;
    const int wave = threadIdx.x >> 5;
    const int tile = blockIdx.x * 4 + wave; // 0 .. B*NH*(L/32)-1
    const int qt = tile & 63;               // 32-query tile index
    const int bh = tile >> 6;               // b*NH + h
    const int half = lane >> 4, cn = lane & 15;
    const int rowlane = 8 * half;           // base lane for row-space bcast

    const __bf16* Qb = Q + (size_t)bh * SEQL * DH + (size_t)qt * 32 * DH;
    const __bf16* Kb = Kd + (size_t)bh * SEQL * DH;
    const __bf16* Vb = Vt + (size_t)bh * DH * SEQL;

    // Q as B-fragments (column n of Q^T = contiguous row of Q); two 16-query
    // tiles x two 32-dim halves. Loaded once.
    v16bf bQ[2][2];
    bQ[0][0] = load_b_frag(Qb, DH, lane);
    bQ[0][1] = load_b_frag(Qb + 32, DH, lane);
    bQ[1][0] = load_b_frag(Qb + 16 * DH, DH, lane);
    bQ[1][1] = load_b_frag(Qb + 16 * DH + 32, DH, lane);

    float rm[2] = {-3.0e38f, -3.0e38f};
    float rl[2] = {0.0f, 0.0f};
    v8f o[2][4];
#pragma unroll
    for (int t = 0; t < 2; ++t)
#pragma unroll
        for (int d = 0; d < 4; ++d) o[t][d] = (v8f){};

    for (int kb = 0; kb < SEQL; kb += 32) {
        // K rows as A-fragments: keys (kb..kb+15) and (kb+16..kb+31),
        // dims split 0..31 / 32..63.
        v16bf aK0a = load_a_frag(Kb + (size_t)kb * DH, DH, lane);
        v16bf aK0b = load_a_frag(Kb + (size_t)kb * DH + 32, DH, lane);
        v16bf aK1a = load_a_frag(Kb + (size_t)(kb + 16) * DH, DH, lane);
        v16bf aK1b = load_a_frag(Kb + (size_t)(kb + 16) * DH + 32, DH, lane);

        // V as B-fragments (Vt is dim-major -> contiguous), shared by both
        // query tiles.
        v16bf bv0 = load_b_frag(Vb + (size_t)0 * SEQL + kb, SEQL, lane);
        v16bf bv1 = load_b_frag(Vb + (size_t)16 * SEQL + kb, SEQL, lane);
        v16bf bv2 = load_b_frag(Vb + (size_t)32 * SEQL + kb, SEQL, lane);
        v16bf bv3 = load_b_frag(Vb + (size_t)48 * SEQL + kb, SEQL, lane);

#pragma unroll
        for (int t = 0; t < 2; ++t) {
            // S^T tiles: st0 = keys kb..kb+15, st1 = keys kb+16..kb+31,
            // N = 16 queries of tile t. Lane holds query cn, keys
            // {8h+i} u {16+8h+i}.
            v8f st0 = {}, st1 = {};
            st0 = wmma_bf16(aK0a, bQ[t][0], st0);
            st0 = wmma_bf16(aK0b, bQ[t][1], st0);
            st1 = wmma_bf16(aK1a, bQ[t][0], st1);
            st1 = wmma_bf16(aK1b, bQ[t][1], st1);

            // --- per-query online softmax (in-lane + one xor-16) ---
            float mx = st0[0];
#pragma unroll
            for (int i = 1; i < 8; ++i) mx = fmaxf(mx, st0[i]);
#pragma unroll
            for (int i = 0; i < 8; ++i) mx = fmaxf(mx, st1[i]);
            mx = fmaxf(mx, __shfl_xor(mx, 16));
            float nm = fmaxf(rm[t], mx);
            float corr = __expf(rm[t] - nm);
            rm[t] = nm;

            // exp() straight into the A-fragment of P (layout identity).
            v16bf ap;
            float sum = 0.0f;
#pragma unroll
            for (int i = 0; i < 8; ++i) {
                float e = __expf(st0[i] - nm);
                sum += e;
                ap[i] = (__bf16)e;
            }
#pragma unroll
            for (int i = 0; i < 8; ++i) {
                float e = __expf(st1[i] - nm);
                sum += e;
                ap[i + 8] = (__bf16)e;
            }
            sum += __shfl_xor(sum, 16);
            rl[t] = rl[t] * corr + sum;

            // Broadcast corr from query-lane space to O row space.
            float cr[8];
#pragma unroll
            for (int i = 0; i < 8; ++i) cr[i] = __shfl(corr, rowlane + i);
#pragma unroll
            for (int i = 0; i < 8; ++i) {
                o[t][0][i] *= cr[i];
                o[t][1][i] *= cr[i];
                o[t][2][i] *= cr[i];
                o[t][3][i] *= cr[i];
            }

            // O += P @ V
            o[t][0] = wmma_bf16(ap, bv0, o[t][0]);
            o[t][1] = wmma_bf16(ap, bv1, o[t][1]);
            o[t][2] = wmma_bf16(ap, bv2, o[t][2]);
            o[t][3] = wmma_bf16(ap, bv3, o[t][3]);
        }
    }

    // ---- normalize and store: attn_out[b, l, h*64 + d] ----
    const int b = bh / NH, hh = bh % NH;
#pragma unroll
    for (int t = 0; t < 2; ++t) {
        float inv = 1.0f / rl[t];
        float ir[8];
#pragma unroll
        for (int i = 0; i < 8; ++i) ir[i] = __shfl(inv, rowlane + i);
#pragma unroll
        for (int i = 0; i < 8; ++i) {
            int l = qt * 32 + t * 16 + i + 8 * half;
            __bf16* dst = Oa + ((size_t)b * SEQL + l) * HID + hh * 64 + cn;
            dst[0]  = (__bf16)(o[t][0][i] * ir[i]);
            dst[16] = (__bf16)(o[t][1][i] * ir[i]);
            dst[32] = (__bf16)(o[t][2][i] * ir[i]);
            dst[48] = (__bf16)(o[t][3][i] * ir[i]);
        }
    }
}

// ---------------------------------------------------------------------------
// Launch
// ---------------------------------------------------------------------------
extern "C" void kernel_launch(void* const* d_in, const int* in_sizes, int n_in,
                              void* d_out, int out_size, void* d_ws,
                              size_t ws_size, hipStream_t stream) {
    const float* x    = (const float*)d_in[0]; // [2,2048,1536]
    const float* wqkv = (const float*)d_in[1]; // [4608,1536]
    const float* wo   = (const float*)d_in[2]; // [1536,1536]
    const float* cosb = (const float*)d_in[3]; // [2048,64]
    const float* sinb = (const float*)d_in[4]; // [2048,64]
    float* out = (float*)d_out;                // [2,2048,1536]

    char* ws = (char*)d_ws;
    size_t off = 0;
    auto alloc = [&](size_t bytes) -> void* {
        void* p = ws + off;
        off = (off + bytes + 255) & ~(size_t)255;
        return p;
    };
    __bf16* xb    = (__bf16*)alloc((size_t)MROWS * HID * 2);
    __bf16* wqkvb = (__bf16*)alloc((size_t)EQKV * HID * 2);
    __bf16* wob   = (__bf16*)alloc((size_t)HID * HID * 2);
    __bf16* qkvb  = (__bf16*)alloc((size_t)MROWS * EQKV * 2);
    __bf16* Qb    = (__bf16*)alloc((size_t)BATCH * NH * SEQL * DH * 2);
    __bf16* Kb    = (__bf16*)alloc((size_t)BATCH * NH * SEQL * DH * 2);
    __bf16* Vtb   = (__bf16*)alloc((size_t)BATCH * NH * SEQL * DH * 2);
    __bf16* attnb = (__bf16*)alloc((size_t)MROWS * HID * 2);

    // 1) fp32 -> bf16 casts (vectorized x4)
    int nx = MROWS * HID / 4, nwq = EQKV * HID / 4, nwo = HID * HID / 4;
    cvt_bf16x4_kernel<<<(nx + 255) / 256, 256, 0, stream>>>(x, xb, nx);
    cvt_bf16x4_kernel<<<(nwq + 255) / 256, 256, 0, stream>>>(wqkv, wqkvb, nwq);
    cvt_bf16x4_kernel<<<(nwo + 255) / 256, 256, 0, stream>>>(wo, wob, nwo);

    // 2) QKV projection: [4096,1536] x [4608,1536]^T -> [4096,4608] (bf16)
    gemm_bf16_kernel<true><<<dim3(EQKV / 128, MROWS / 128), 256, 0, stream>>>(
        xb, wqkvb, qkvb, MROWS, EQKV, HID);

    // 3) RoPE + split into Q (scaled), K, V^T
    int nrope = BATCH * SEQL * NH * DH;
    rope_split_kernel<<<nrope / 256, 256, 0, stream>>>(qkvb, cosb, sinb, Qb,
                                                       Kb, Vtb);

    // 4) Flash attention: 3072 (b,h, 32-query) waves, 4 waves/block
    flash_attn_kernel<<<(BATCH * NH * (SEQL / 32)) / 4, 128, 0, stream>>>(
        Qb, Kb, Vtb, attnb);

    // 5) Output projection: [4096,1536] x [1536,1536]^T -> out (fp32)
    gemm_bf16_kernel<false><<<dim3(HID / 128, MROWS / 128), 256, 0, stream>>>(
        attnb, wob, out, MROWS, HID, HID);
}
